// Classifier_86260123173820
// MI455X (gfx1250) — compile-verified, matrix operand order checked by reading, not run on hardware
//
#include <hip/hip_runtime.h>
#include <hip/hip_bf16.h>
#include <math.h>

// ---------------- problem constants ----------------
#define NODES   131072      // B*NPG
#define EDGES   2097152     // NODES*DEG
#define BGR     64
#define NPG     2048
#define NF      128
#define EF      32
#define IN0     160         // NF+EF
#define HIDL    32
#define TLDIM   97
#define TOPK    30
#define C1C     16
#define C2C     32
#define KW2C    5
#define DENSE   352
#define HIDD    128
#define NCLS    10

typedef __attribute__((ext_vector_type(16))) __bf16 v16bf;
typedef __attribute__((ext_vector_type(8)))  __bf16 v8bf;
typedef __attribute__((ext_vector_type(8)))  float  v8f;

__device__ __forceinline__ unsigned short f2bf(float f) {
    unsigned int u = __float_as_uint(f);
    u += 0x7FFFu + ((u >> 16) & 1u);          // round-to-nearest-even
    return (unsigned short)(u >> 16);
}

// concat(node_feat, e2n) feature accessor for layer 0
__device__ __forceinline__ float feat0(const float* nf, const float* e2n, int n, int c) {
    return (c < NF) ? nf[n * NF + c] : e2n[n * EF + (c - NF)];
}

// ---------------- scatter / gather kernels ----------------

// e2n = segment_sum(edge_feat, edge_row)
__global__ void k_scatter_e2n(const float* __restrict__ ef, const int* __restrict__ row,
                              float* __restrict__ e2n) {
    int t = blockIdx.x * blockDim.x + threadIdx.x;
    if (t >= EDGES * EF) return;
    int e = t >> 5;          // /32 channels
    int c = t & 31;
    atomicAdd(&e2n[row[e] * EF + c], ef[t]);
}

// pooled[n, 0:160] = concat(node_feat, e2n)[n]   (residual term)
__global__ void k_init_pooled_l0(const float* __restrict__ nf, const float* __restrict__ e2n,
                                 float* __restrict__ pooled) {
    int t = blockIdx.x * blockDim.x + threadIdx.x;
    if (t >= NODES * IN0) return;
    int n = t / IN0, c = t - n * IN0;
    pooled[t] = feat0(nf, e2n, n, c);
}

// pooled[row[e]] += concat_feat[col[e]]
__global__ void k_scatter_l0(const float* __restrict__ nf, const float* __restrict__ e2n,
                             const int* __restrict__ row, const int* __restrict__ col,
                             float* __restrict__ pooled) {
    int t = blockIdx.x * blockDim.x + threadIdx.x;
    if (t >= EDGES * IN0) return;
    int e = t / IN0, c = t - e * IN0;
    atomicAdd(&pooled[row[e] * IN0 + c], feat0(nf, e2n, col[e], c));
}

// pooled[n, 0:32] = cat[n, off:off+32]   (residual term, layers 1..3)
__global__ void k_init_pooled_cat(const float* __restrict__ cat, int off,
                                  float* __restrict__ pooled) {
    int t = blockIdx.x * blockDim.x + threadIdx.x;
    if (t >= NODES * HIDL) return;
    int n = t >> 5, c = t & 31;
    pooled[n * IN0 + c] = cat[n * TLDIM + off + c];
}

__global__ void k_scatter_cat(const float* __restrict__ cat, int off,
                              const int* __restrict__ row, const int* __restrict__ col,
                              float* __restrict__ pooled) {
    int t = blockIdx.x * blockDim.x + threadIdx.x;
    if (t >= EDGES * HIDL) return;
    int e = t >> 5, c = t & 31;
    atomicAdd(&pooled[row[e] * IN0 + c], cat[col[e] * TLDIM + off + c]);
}

// ---------------- f32 -> bf16 staging ----------------

__global__ void k_cvt_pooled(const float* __restrict__ src, unsigned short* __restrict__ dst,
                             int width, int total) {
    int t = blockIdx.x * blockDim.x + threadIdx.x;
    if (t >= total) return;
    int n = t / width, c = t - n * width;
    dst[n * IN0 + c] = f2bf(src[n * IN0 + c]);
}

__global__ void k_cvt_w(const float* __restrict__ W, unsigned short* __restrict__ dst, int count) {
    int t = blockIdx.x * blockDim.x + threadIdx.x;
    if (t < count) dst[t] = f2bf(W[t]);
}

// ---------------- WMMA GEMM + bias + degnorm + tanh ----------------
// out[n, o] = tanh((sum_k pooled[n,k] * W[o,k] + b[o]) / deg[n])
// One wave32 computes a 16x16 output tile via v_wmma_f32_16x16x32_bf16.
// A (16x32 bf16): lane L<16 -> row M=L, bf16 pairs K={0..7,16..23}; lanes 16..31 -> K offset +8.
// B (32x16 bf16): symmetric with N = lane&15 (B column o = W row o).
// C/D (16x16 f32): VGPR r: lanes 0-15 -> M=r, lanes 16-31 -> M=r+8; N = lane&15.
__global__ void k_gnn_gemm(const unsigned short* __restrict__ Au,
                           const unsigned short* __restrict__ Wu,
                           const float* __restrict__ bias, const float* __restrict__ degs,
                           float* __restrict__ cat, int col_off,
                           int d_in, int out_cols, int n_tiles) {
    const __bf16* A = reinterpret_cast<const __bf16*>(Au);
    const __bf16* W = reinterpret_cast<const __bf16*>(Wu);
    int wave = (int)((blockIdx.x * blockDim.x + threadIdx.x) >> 5);
    int lane = threadIdx.x & 31;
    int tm = wave / n_tiles;
    int tn = wave - tm * n_tiles;
    if (tm >= NODES / 16) return;
    int half = lane >> 4;       // 0: lanes 0-15, 1: lanes 16-31
    int l15  = lane & 15;
    int arow = tm * 16 + l15;   // node row
    int brow = tn * 16 + l15;   // weight row (zero-padded past out_cols)

    v8f acc = {};
    for (int kc = 0; kc < d_in; kc += 32) {
        const __bf16* pa = A + arow * IN0 + kc + half * 8;
        v8bf alo = *reinterpret_cast<const v8bf*>(pa);
        v8bf ahi = *reinterpret_cast<const v8bf*>(pa + 16);
        const __bf16* pb = W + brow * d_in + kc + half * 8;
        v8bf blo = *reinterpret_cast<const v8bf*>(pb);
        v8bf bhi = *reinterpret_cast<const v8bf*>(pb + 16);
        v16bf a, b;
#pragma unroll
        for (int i = 0; i < 8; ++i) {
            a[i] = alo[i]; a[8 + i] = ahi[i];
            b[i] = blo[i]; b[8 + i] = bhi[i];
        }
        acc = __builtin_amdgcn_wmma_f32_16x16x32_bf16(
            /*neg_a=*/false, a, /*neg_b=*/false, b,
            /*c_mod=*/(short)0, acc, /*reuse_a=*/false, /*reuse_b=*/false);
    }

    int ncol = tn * 16 + l15;
    if (ncol < out_cols) {
        float bv = bias[ncol];
#pragma unroll
        for (int r = 0; r < 8; ++r) {
            int m = tm * 16 + r + half * 8;
            float v = tanhf((acc[r] + bv) / degs[m]);
            cat[m * TLDIM + col_off + ncol] = v;
        }
    }
}

// ---------------- SortPooling + CNN head, one block per graph ----------------
__global__ void k_head(const float* __restrict__ cat,
                       const float* __restrict__ Wc1, const float* __restrict__ bc1,
                       const float* __restrict__ Wc2, const float* __restrict__ bc2,
                       const float* __restrict__ Wh,  const float* __restrict__ bh,
                       const float* __restrict__ Wo,  const float* __restrict__ bo,
                       float* __restrict__ out) {
    __shared__ float sv[NPG];
    __shared__ float rv[256];
    __shared__ int   ri[256];
    __shared__ int   sel[TOPK];
    __shared__ float pg[TOPK * TLDIM];
    __shared__ float y1[C1C * TOPK];
    __shared__ float y2[C1C * (TOPK / 2)];
    __shared__ float d3[DENSE];
    __shared__ float h1[HIDD];
    __shared__ float lg[NCLS];

    int g = blockIdx.x, tid = threadIdx.x;

    for (int t = tid; t < NPG; t += 256) sv[t] = cat[(g * NPG + t) * TLDIM + (TLDIM - 1)];
    __syncthreads();

    // top-K by iterative argmax (value desc, ties -> lower index, matching lax.top_k)
    for (int it = 0; it < TOPK; ++it) {
        float best = -INFINITY; int bi = NPG;
        for (int t = tid; t < NPG; t += 256) {
            float v = sv[t];
            if (v > best || (v == best && t < bi)) { best = v; bi = t; }
        }
        rv[tid] = best; ri[tid] = bi;
        __syncthreads();
        for (int s = 128; s > 0; s >>= 1) {
            if (tid < s) {
                float vo = rv[tid + s]; int io = ri[tid + s];
                if (vo > rv[tid] || (vo == rv[tid] && io < ri[tid])) { rv[tid] = vo; ri[tid] = io; }
            }
            __syncthreads();
        }
        if (tid == 0) { sel[it] = ri[0]; sv[ri[0]] = -INFINITY; }
        __syncthreads();
    }

    // gather 30x97 features
    for (int t = tid; t < TOPK * TLDIM; t += 256) {
        int j = t / TLDIM, dd = t - j * TLDIM;
        pg[t] = cat[(g * NPG + sel[j]) * TLDIM + dd];
    }
    __syncthreads();

    // conv1 (per-node linear 97->16) + relu
    for (int t = tid; t < C1C * TOPK; t += 256) {
        int c = t / TOPK, k = t - c * TOPK;
        float s = bc1[c];
        for (int dd = 0; dd < TLDIM; ++dd) s += pg[k * TLDIM + dd] * Wc1[c * TLDIM + dd];
        y1[c * TOPK + k] = fmaxf(s, 0.f);
    }
    __syncthreads();

    // maxpool1d(2,2): 30 -> 15
    for (int t = tid; t < C1C * 15; t += 256) {
        int c = t / 15, p = t - c * 15;
        y2[c * 15 + p] = fmaxf(y1[c * TOPK + 2 * p], y1[c * TOPK + 2 * p + 1]);
    }
    __syncthreads();

    // conv1d 16->32 kw=5, width 15 -> 11, relu (relu(relu(.)) == relu)
    for (int t = tid; t < C2C * 11; t += 256) {
        int o = t / 11, x = t - o * 11;
        float s = bc2[o];
        for (int i = 0; i < C1C; ++i)
            for (int w = 0; w < KW2C; ++w)
                s += y2[i * 15 + x + w] * Wc2[(o * C1C + i) * KW2C + w];
        d3[o * 11 + x] = fmaxf(s, 0.f);
    }
    __syncthreads();

    // dense 352 -> 128, relu
    for (int t = tid; t < HIDD; t += 256) {
        float s = bh[t];
        for (int dd = 0; dd < DENSE; ++dd) s += d3[dd] * Wh[t * DENSE + dd];
        h1[t] = fmaxf(s, 0.f);
    }
    __syncthreads();

    // dense 128 -> 10 + log_softmax
    if (tid < NCLS) {
        float s = bo[tid];
        for (int j = 0; j < HIDD; ++j) s += h1[j] * Wo[tid * HIDD + j];
        lg[tid] = s;
    }
    __syncthreads();
    if (tid < NCLS) {
        float m = lg[0];
        for (int i = 1; i < NCLS; ++i) m = fmaxf(m, lg[i]);
        float se = 0.f;
        for (int i = 0; i < NCLS; ++i) se += expf(lg[i] - m);
        out[g * NCLS + tid] = lg[tid] - m - logf(se);
    }
}

// ---------------- launch ----------------
extern "C" void kernel_launch(void* const* d_in, const int* in_sizes, int n_in,
                              void* d_out, int out_size, void* d_ws, size_t ws_size,
                              hipStream_t stream) {
    (void)in_sizes; (void)n_in; (void)out_size; (void)ws_size;

    const float* node_feat = (const float*)d_in[0];
    const float* edge_feat = (const float*)d_in[1];
    const float* degs      = (const float*)d_in[2];
    const float* Wl[4]     = {(const float*)d_in[3], (const float*)d_in[5],
                              (const float*)d_in[7], (const float*)d_in[9]};
    const float* bl[4]     = {(const float*)d_in[4], (const float*)d_in[6],
                              (const float*)d_in[8], (const float*)d_in[10]};
    const float* Wc1 = (const float*)d_in[11];
    const float* bc1 = (const float*)d_in[12];
    const float* Wc2 = (const float*)d_in[13];
    const float* bc2 = (const float*)d_in[14];
    const float* Wh  = (const float*)d_in[15];
    const float* bh  = (const float*)d_in[16];
    const float* Wo  = (const float*)d_in[17];
    const float* bo  = (const float*)d_in[18];
    const int* erow  = (const int*)d_in[19];
    const int* ecol  = (const int*)d_in[20];
    float* out = (float*)d_out;

    // workspace layout
    char* ws = (char*)d_ws;
    float*          pooled = (float*)ws;                                   // N*160 f32
    unsigned short* pbf    = (unsigned short*)(ws + (size_t)NODES*IN0*4);  // N*160 bf16
    float*          e2n    = (float*)((char*)pbf + (size_t)NODES*IN0*2);   // N*32 f32
    float*          cat    = (float*)((char*)e2n + (size_t)NODES*EF*4);    // N*97 f32
    unsigned short* wbf    = (unsigned short*)((char*)cat + (size_t)NODES*TLDIM*4); // 32*160 bf16

    // ---- e2n segment sum ----
    hipMemsetAsync(e2n, 0, (size_t)NODES * EF * 4, stream);
    k_scatter_e2n<<<(EDGES * EF) / 256, 256, 0, stream>>>(edge_feat, erow, e2n);

    // ---- layer 0 (d_in=160 -> 32) ----
    k_init_pooled_l0<<<(NODES * IN0) / 256, 256, 0, stream>>>(node_feat, e2n, pooled);
    k_scatter_l0<<<(unsigned)(((size_t)EDGES * IN0) / 256), 256, 0, stream>>>(
        node_feat, e2n, erow, ecol, pooled);
    hipMemsetAsync(wbf, 0, 32 * 160 * 2, stream);
    k_cvt_w<<<(32 * 160 + 255) / 256, 256, 0, stream>>>(Wl[0], wbf, 32 * 160);
    k_cvt_pooled<<<(NODES * IN0) / 256, 256, 0, stream>>>(pooled, pbf, IN0, NODES * IN0);
    k_gnn_gemm<<<(NODES / 16) * 2 / 8, 256, 0, stream>>>(
        pbf, wbf, bl[0], degs, cat, /*col_off=*/0, /*d_in=*/IN0, /*out_cols=*/32, /*n_tiles=*/2);

    // ---- layers 1..3 (d_in=32 -> {32,32,1}) ----
    for (int l = 1; l < 4; ++l) {
        int off_in  = 32 * (l - 1);
        int off_out = 32 * l;
        int out_cols = (l == 3) ? 1 : 32;
        int ntiles   = (l == 3) ? 1 : 2;

        k_init_pooled_cat<<<(NODES * HIDL) / 256, 256, 0, stream>>>(cat, off_in, pooled);
        k_scatter_cat<<<(EDGES * HIDL) / 256, 256, 0, stream>>>(cat, off_in, erow, ecol, pooled);
        hipMemsetAsync(wbf, 0, 32 * 160 * 2, stream);
        k_cvt_w<<<(out_cols * HIDL + 255) / 256, 256, 0, stream>>>(Wl[l], wbf, out_cols * HIDL);
        k_cvt_pooled<<<(NODES * HIDL) / 256, 256, 0, stream>>>(pooled, pbf, HIDL, NODES * HIDL);
        k_gnn_gemm<<<(NODES / 16) * ntiles / 8, 256, 0, stream>>>(
            pbf, wbf, bl[l], degs, cat, off_out, HIDL, out_cols, ntiles);
    }

    // ---- sortpool + CNN head ----
    k_head<<<BGR, 256, 0, stream>>>(cat, Wc1, bc1, Wc2, bc2, Wh, bh, Wo, bo, out);
}
